// Encoder_27642409517668
// MI455X (gfx1250) — matrix operand order repaired and checked by reading, compiled, not run on hardware
//
#include <hip/hip_runtime.h>
#include <hip/hip_bf16.h>

// ---------------------------------------------------------------------------
// Transformer encoder for MI455X (gfx1250): all GEMMs + attention on
// v_wmma_f32_16x16x32_f16 (wave32). Weights/activations converted to f16 with
// f32 accumulation; softmax/layernorm in f32.
// ---------------------------------------------------------------------------

typedef __attribute__((ext_vector_type(16))) _Float16 v16h;
typedef __attribute__((ext_vector_type(8)))  _Float16 v8h;
typedef __attribute__((ext_vector_type(8)))  float    v8f;

#define DEVINL __device__ __forceinline__

static constexpr int NB   = 8;     // batch
static constexpr int SEQ  = 1024;  // sequence
static constexpr int FIN  = 64;    // input features
static constexpr int EMB  = 512;   // embed
static constexpr int HEADS= 8;
static constexpr int HDIM = 64;
static constexpr int OUTC = 64;
static constexpr int LAY  = 6;
static constexpr int FFD  = 2048;
static constexpr int NS   = NB * SEQ; // 8192 rows

DEVINL v16h cat16(v8h lo, v8h hi) {
    return __builtin_shufflevector(lo, hi, 0,1,2,3,4,5,6,7,8,9,10,11,12,13,14,15);
}

DEVINL v8f wmma_f16(v16h a, v16h b, v8f c) {
    return __builtin_amdgcn_wmma_f32_16x16x32_f16(false, a, false, b, (short)0, c, false, false);
}

// A fragment 16x32 (MxK), f16. Abase already offset to tile row 0.
// Lane<16: row=lane, K {k0..k0+7, k0+16..k0+23}; lane>=16: row=lane-16, K {+8..+15, +24..+31}.
DEVINL v16h load_a_frag(const _Float16* __restrict__ Abase, int lda, int lane, int k0) {
    int row = lane & 15;
    int kb  = k0 + ((lane & 16) ? 8 : 0);
    const _Float16* p = Abase + (size_t)row * lda;
    v8h lo = *(const v8h*)(p + kb);
    v8h hi = *(const v8h*)(p + kb + 16);
    return cat16(lo, hi);
}

// B fragment 32x16 (KxN) loaded from transposed weight Bt[n][k] (K-contiguous).
// Bt already offset to column tile 0. Lane<16: col=lane, K {k0..k0+7, k0+16..k0+23}; lane>=16: K {+8..+15, +24..+31}.
DEVINL v16h load_b_frag(const _Float16* __restrict__ Btbase, int ldb, int lane, int k0) {
    int col = lane & 15;
    int kb  = k0 + ((lane & 16) ? 8 : 0);
    const _Float16* p = Btbase + (size_t)col * ldb + kb;
    v8h lo = *(const v8h*)(p);
    v8h hi = *(const v8h*)(p + 16);
    return cat16(lo, hi);
}

// ---------------------------------------------------------------------------
// Prep kernels
// ---------------------------------------------------------------------------

// dst[b][c][r] = (f16) src[b][r][c]
__global__ void k_transpose_f32f16(const float* __restrict__ src, _Float16* __restrict__ dst,
                                   int B, int R, int C) {
    long idx = (long)blockIdx.x * blockDim.x + threadIdx.x;
    long total = (long)B * R * C;
    if (idx >= total) return;
    long rc = (long)R * C;
    int b = (int)(idx / rc);
    int rem = (int)(idx - (long)b * rc);
    int r = rem / C, c = rem % C;
    dst[(long)b * rc + (long)c * R + r] = (_Float16)src[idx];
}

// Wsrc: (LAY, 64, 64) [d][e]; dst: (LAY, 512, 512) block-diag transposed: dst[l][eo][ei]
__global__ void k_expand_blockdiag(const float* __restrict__ W, _Float16* __restrict__ dst) {
    long idx = (long)blockIdx.x * blockDim.x + threadIdx.x;
    long total = (long)LAY * EMB * EMB;
    if (idx >= total) return;
    int l  = (int)(idx / (EMB * EMB));
    int rem = (int)(idx - (long)l * EMB * EMB);
    int eo = rem / EMB, ei = rem % EMB;
    _Float16 v = (_Float16)0.0f;
    if ((eo >> 6) == (ei >> 6))
        v = (_Float16)W[(long)l * 4096 + (ei & 63) * 64 + (eo & 63)];
    dst[idx] = v;
}

// v: (N,S,H,64) f16 -> vt: (N,H,64,S) f16
__global__ void k_vtranspose(const _Float16* __restrict__ v, _Float16* __restrict__ vt) {
    long idx = (long)blockIdx.x * blockDim.x + threadIdx.x;
    if (idx >= (long)NS * EMB) return;
    int d = idx & 63;
    long t = idx >> 6;
    int h = (int)(t & (HEADS - 1)); t >>= 3;
    int s = (int)(t & (SEQ - 1));
    int n = (int)(t >> 10);
    vt[(((long)(n * HEADS + h) * HDIM + d) << 10) + s] = v[idx];
}

// ---------------------------------------------------------------------------
// Generic WMMA GEMM: C(MxN) = A(MxK) * Bt(NxK)^T, fused epilogues.
// Wave tile 16x64 (4 accumulators); block = 8 waves = 32x256 tile.
// mode 0: +bias -> f16        mode 1: relu(+bias) -> f16
// mode 2: relu(+bias)+posemb -> f16
// mode 3: +bias+resid16 -> f32
// mode 4: +bias, transposed store out[m/S][c][m%S] (f32)
// ---------------------------------------------------------------------------
__global__ __launch_bounds__(256) void k_gemm_wmma(
    const _Float16* __restrict__ A, int lda,
    const _Float16* __restrict__ Bt, int ldb,
    const float* __restrict__ bias, int bias_mod,
    int M, int K, int N, int mode,
    _Float16* __restrict__ out16, float* __restrict__ out32,
    const float* __restrict__ posemb, const _Float16* __restrict__ resid16,
    int Sdim)
{
    const int lane = threadIdx.x & 31;
    const int wave = threadIdx.x >> 5;
    const int m0 = (blockIdx.x * 2 + (wave & 1)) * 16;
    const int n0 = (blockIdx.y * 4 + (wave >> 1)) * 64;
    if (m0 + 16 > M || n0 + 64 > N) return;

    const v8f z8 = {0.f,0.f,0.f,0.f,0.f,0.f,0.f,0.f};
    v8f acc[4] = {z8, z8, z8, z8};

    const _Float16* Abase = A + (size_t)m0 * lda;
    const _Float16* Bbase0 = Bt + (size_t)(n0 +  0) * ldb;
    const _Float16* Bbase1 = Bt + (size_t)(n0 + 16) * ldb;
    const _Float16* Bbase2 = Bt + (size_t)(n0 + 32) * ldb;
    const _Float16* Bbase3 = Bt + (size_t)(n0 + 48) * ldb;

    for (int k0 = 0; k0 < K; k0 += 32) {
        v16h a = load_a_frag(Abase, lda, lane, k0);
        acc[0] = wmma_f16(a, load_b_frag(Bbase0, ldb, lane, k0), acc[0]);
        acc[1] = wmma_f16(a, load_b_frag(Bbase1, ldb, lane, k0), acc[1]);
        acc[2] = wmma_f16(a, load_b_frag(Bbase2, ldb, lane, k0), acc[2]);
        acc[3] = wmma_f16(a, load_b_frag(Bbase3, ldb, lane, k0), acc[3]);
    }

    const int colBase = lane & 15;
    const int rowOff  = (lane & 16) ? 8 : 0;
    #pragma unroll
    for (int t = 0; t < 4; t++) {
        int c = n0 + t * 16 + colBase;
        float bv = bias ? bias[c % bias_mod] : 0.f;
        #pragma unroll
        for (int r = 0; r < 8; r++) {
            int m = m0 + rowOff + r;
            float v = acc[t][r] + bv;
            if (mode == 1 || mode == 2) v = fmaxf(v, 0.f);
            if (mode == 2) v += posemb[(size_t)(m % Sdim) * N + c];
            if (mode == 3) v += (float)resid16[(size_t)m * N + c];
            if (mode == 4) {
                int n = m / Sdim, s = m % Sdim;
                out32[((size_t)n * N + c) * Sdim + s] = v;
            } else if (mode == 3) {
                out32[(size_t)m * N + c] = v;
            } else {
                out16[(size_t)m * N + c] = (_Float16)v;
            }
        }
    }
}

// ---------------------------------------------------------------------------
// Flash attention: per wave one (n,h,16-row q tile); 32-wide kv blocks.
// q,k,o: (N,S,H,64) f16 (row stride 512); vt: (N,H,64,S) f16 (row stride S).
// ---------------------------------------------------------------------------
__global__ __launch_bounds__(256) void k_flash_attn(
    const _Float16* __restrict__ q, const _Float16* __restrict__ k,
    const _Float16* __restrict__ vt, _Float16* __restrict__ o)
{
    __shared__ _Float16 lds[8][16][32];   // per-wave 16x32 P tile
    const int lane = threadIdx.x & 31;
    const int wave = threadIdx.x >> 5;
    const int w  = blockIdx.x * 8 + wave;
    const int qb = w & 63;             // q tile within sequence (S/16 = 64)
    const int nh = w >> 6;
    const int n  = nh >> 3, h = nh & 7;

    const _Float16* qbase = q  + ((size_t)(n * SEQ + qb * 16) * HEADS + h) * HDIM;
    const _Float16* kbase = k  + ((size_t)(n * SEQ) * HEADS + h) * HDIM;
    const _Float16* vbase = vt + ((size_t)(n * HEADS + h) * HDIM) * (size_t)SEQ;

    v16h qa0 = load_a_frag(qbase, EMB, lane, 0);
    v16h qa1 = load_a_frag(qbase, EMB, lane, 32);

    const v8f z8 = {0.f,0.f,0.f,0.f,0.f,0.f,0.f,0.f};
    v8f acc[4] = {z8, z8, z8, z8};
    float mrun[8], lrun[8];
    #pragma unroll
    for (int r = 0; r < 8; r++) { mrun[r] = -1e30f; lrun[r] = 0.f; }

    const float inv_scale = 0.044194173824159216f; // 1/sqrt(512)
    const int colBase = lane & 15;
    const int rowOff  = (lane & 16) ? 8 : 0;

    for (int j = 0; j < SEQ; j += 32) {
        // scores: Q(16x64) @ K^T -> 16x32
        v8f sc0 = z8, sc1 = z8;
        const _Float16* kb0 = kbase + (size_t)j * EMB;
        const _Float16* kb1 = kbase + (size_t)(j + 16) * EMB;
        sc0 = wmma_f16(qa0, load_b_frag(kb0, EMB, lane, 0),  sc0);
        sc0 = wmma_f16(qa1, load_b_frag(kb0, EMB, lane, 32), sc0);
        sc1 = wmma_f16(qa0, load_b_frag(kb1, EMB, lane, 0),  sc1);
        sc1 = wmma_f16(qa1, load_b_frag(kb1, EMB, lane, 32), sc1);

        // online softmax over the 32 columns (row = C-frag vgpr index; cols = lanes)
        #pragma unroll
        for (int r = 0; r < 8; r++) {
            float v0 = sc0[r] * inv_scale, v1 = sc1[r] * inv_scale;
            float bm = fmaxf(v0, v1);
            bm = fmaxf(bm, __shfl_xor(bm, 1, 32));
            bm = fmaxf(bm, __shfl_xor(bm, 2, 32));
            bm = fmaxf(bm, __shfl_xor(bm, 4, 32));
            bm = fmaxf(bm, __shfl_xor(bm, 8, 32));
            float mnew  = fmaxf(mrun[r], bm);
            float alpha = __expf(mrun[r] - mnew);
            mrun[r] = mnew;
            float p0 = __expf(v0 - mnew), p1 = __expf(v1 - mnew);
            sc0[r] = p0; sc1[r] = p1;
            float rs = p0 + p1;
            rs += __shfl_xor(rs, 1, 32);
            rs += __shfl_xor(rs, 2, 32);
            rs += __shfl_xor(rs, 4, 32);
            rs += __shfl_xor(rs, 8, 32);
            lrun[r] = lrun[r] * alpha + rs;
            #pragma unroll
            for (int t = 0; t < 4; t++) acc[t][r] *= alpha;
        }

        // C-layout -> A-layout for P via LDS (wave-local; DS ops in-order per wave)
        #pragma unroll
        for (int r = 0; r < 8; r++) {
            lds[wave][rowOff + r][colBase]      = (_Float16)sc0[r];
            lds[wave][rowOff + r][16 + colBase] = (_Float16)sc1[r];
        }
        asm volatile("s_wait_dscnt 0" ::: "memory");
        v16h pa;
        {
            int row = lane & 15;
            int kb  = (lane & 16) ? 8 : 0;
            v8h lo = *(const v8h*)&lds[wave][row][kb];
            v8h hi = *(const v8h*)&lds[wave][row][kb + 16];
            pa = cat16(lo, hi);
        }

        // O += P(16x32) @ V(32x64); V^T rows are d (contig over s)
        #pragma unroll
        for (int t = 0; t < 4; t++) {
            v16h vb = load_b_frag(vbase + (size_t)(t * 16) * SEQ + j, SEQ, lane, 0);
            acc[t] = wmma_f16(pa, vb, acc[t]);
        }
        asm volatile("s_wait_dscnt 0" ::: "memory"); // keep next iter's stores behind loads
    }

    // finalize: divide by l, store (n,s,h,d)
    #pragma unroll
    for (int r = 0; r < 8; r++) {
        float invl = 1.f / lrun[r];
        int s = qb * 16 + rowOff + r;
        _Float16* op = o + ((size_t)(n * SEQ + s) * HEADS + h) * HDIM;
        #pragma unroll
        for (int t = 0; t < 4; t++)
            op[t * 16 + colBase] = (_Float16)(acc[t][r] * invl);
    }
}

// ---------------------------------------------------------------------------
// LayerNorm: one wave per row of E=512, f32 in -> f16 out
// ---------------------------------------------------------------------------
__global__ __launch_bounds__(256) void k_layernorm(
    const float* __restrict__ x, const float* __restrict__ g, const float* __restrict__ b,
    _Float16* __restrict__ out, int M)
{
    const int lane = threadIdx.x & 31;
    const int wave = threadIdx.x >> 5;
    const int row = blockIdx.x * 8 + wave;
    if (row >= M) return;
    const float* p = x + (size_t)row * EMB;
    float vals[16];
    float s = 0.f;
    #pragma unroll
    for (int i = 0; i < 16; i++) { vals[i] = p[i * 32 + lane]; s += vals[i]; }
    s += __shfl_xor(s, 1, 32); s += __shfl_xor(s, 2, 32);
    s += __shfl_xor(s, 4, 32); s += __shfl_xor(s, 8, 32); s += __shfl_xor(s, 16, 32);
    float mean = s * (1.f / EMB);
    float vs = 0.f;
    #pragma unroll
    for (int i = 0; i < 16; i++) { float d = vals[i] - mean; vs += d * d; }
    vs += __shfl_xor(vs, 1, 32); vs += __shfl_xor(vs, 2, 32);
    vs += __shfl_xor(vs, 4, 32); vs += __shfl_xor(vs, 8, 32); vs += __shfl_xor(vs, 16, 32);
    float inv = rsqrtf(vs * (1.f / EMB) + 1e-5f);
    _Float16* op = out + (size_t)row * EMB;
    #pragma unroll
    for (int i = 0; i < 16; i++) {
        int c = i * 32 + lane;
        op[c] = (_Float16)((vals[i] - mean) * inv * g[c] + b[c]);
    }
}

// ---------------------------------------------------------------------------
// Host launcher
// ---------------------------------------------------------------------------
static inline long cdivl(long a, long b) { return (a + b - 1) / b; }

extern "C" void kernel_launch(void* const* d_in, const int* in_sizes, int n_in,
                              void* d_out, int out_size, void* d_ws, size_t ws_size,
                              hipStream_t stream) {
    const float* x       = (const float*)d_in[0];
    const float* W_first = (const float*)d_in[1];
    const float* b_first = (const float*)d_in[2];
    const float* pos_emb = (const float*)d_in[3];
    const float* Wq = (const float*)d_in[4];
    const float* bq = (const float*)d_in[5];
    const float* Wk = (const float*)d_in[6];
    const float* bk = (const float*)d_in[7];
    const float* Wv = (const float*)d_in[8];
    const float* bv = (const float*)d_in[9];
    const float* Wo = (const float*)d_in[10];
    const float* bo = (const float*)d_in[11];
    const float* g1 = (const float*)d_in[12];
    const float* be1= (const float*)d_in[13];
    const float* Wf1= (const float*)d_in[14];
    const float* bf1= (const float*)d_in[15];
    const float* Wf2= (const float*)d_in[16];
    const float* bf2= (const float*)d_in[17];
    const float* g2 = (const float*)d_in[18];
    const float* be2= (const float*)d_in[19];
    const float* Wfin=(const float*)d_in[20];
    const float* bfin=(const float*)d_in[21];
    float* out = (float*)d_out;

    // Workspace bump allocator (256B aligned)
    char* wp = (char*)d_ws;
    auto alloc = [&](size_t bytes) -> char* {
        char* r = wp; wp += (bytes + 255) & ~(size_t)255; return r;
    };
    _Float16* x16  = (_Float16*)alloc((size_t)NS * FIN * 2);
    _Float16* h16  = (_Float16*)alloc((size_t)NS * EMB * 2);
    _Float16* q16  = (_Float16*)alloc((size_t)NS * EMB * 2);   // reused as hx16 after attention
    _Float16* k16  = (_Float16*)alloc((size_t)NS * EMB * 2);
    _Float16* v16  = (_Float16*)alloc((size_t)NS * EMB * 2);
    _Float16* vt16 = (_Float16*)alloc((size_t)NS * EMB * 2);
    _Float16* at16 = (_Float16*)alloc((size_t)NS * EMB * 2);
    _Float16* ff16 = (_Float16*)alloc((size_t)NS * FFD * 2);
    float*    xres = (float*)   alloc((size_t)NS * EMB * 4);
    _Float16* WfirstT = (_Float16*)alloc((size_t)EMB * FIN * 2);
    _Float16* WqTf = (_Float16*)alloc((size_t)LAY * EMB * EMB * 2);
    _Float16* WkTf = (_Float16*)alloc((size_t)LAY * EMB * EMB * 2);
    _Float16* WvTf = (_Float16*)alloc((size_t)LAY * EMB * EMB * 2);
    _Float16* WoT  = (_Float16*)alloc((size_t)LAY * EMB * EMB * 2);
    _Float16* Wf1T = (_Float16*)alloc((size_t)LAY * EMB * FFD * 2);
    _Float16* Wf2T = (_Float16*)alloc((size_t)LAY * FFD * EMB * 2);
    _Float16* WfinT= (_Float16*)alloc((size_t)OUTC * EMB * 2);
    _Float16* hx16 = q16;

    auto tr = [&](const float* src, _Float16* dst, int B, int R, int C) {
        long total = (long)B * R * C;
        k_transpose_f32f16<<<dim3((unsigned)cdivl(total, 256)), 256, 0, stream>>>(src, dst, B, R, C);
    };
    // weight & input prep (deterministic, every call)
    tr(x, x16, NB, FIN, SEQ);            // x16: (N,S,F)
    tr(W_first, WfirstT, 1, FIN, EMB);   // (E,F)
    tr(Wo,  WoT,  LAY, EMB, EMB);        // (f,e) per layer
    tr(Wf1, Wf1T, LAY, EMB, FFD);        // (ff,e)
    tr(Wf2, Wf2T, LAY, FFD, EMB);        // (e,ff)
    tr(Wfin, WfinT, 1, EMB, OUTC);       // (o,e)
    {
        long total = (long)LAY * EMB * EMB;
        dim3 g((unsigned)cdivl(total, 256));
        k_expand_blockdiag<<<g, 256, 0, stream>>>(Wq, WqTf);
        k_expand_blockdiag<<<g, 256, 0, stream>>>(Wk, WkTf);
        k_expand_blockdiag<<<g, 256, 0, stream>>>(Wv, WvTf);
    }

    auto gemm = [&](const _Float16* A, int lda, const _Float16* Bt, int ldb,
                    const float* bias, int bias_mod, int M, int K, int Nn, int mode,
                    _Float16* o16, float* o32, const float* pe, const _Float16* res) {
        dim3 g((unsigned)cdivl(M, 32), (unsigned)cdivl(Nn, 256));
        k_gemm_wmma<<<g, 256, 0, stream>>>(A, lda, Bt, ldb, bias, bias_mod,
                                           M, K, Nn, mode, o16, o32, pe, res, SEQ);
    };

    // h = relu(x @ W_first + b_first) + pos_emb
    gemm(x16, FIN, WfirstT, FIN, b_first, EMB, NS, FIN, EMB, 2, h16, nullptr, pos_emb, nullptr);

    const long vt_total = (long)NS * EMB;
    for (int l = 0; l < LAY; l++) {
        // QKV (block-diagonal 512x512 per-head projection)
        gemm(h16, EMB, WqTf + (size_t)l * EMB * EMB, EMB, bq + l * HDIM, HDIM,
             NS, EMB, EMB, 0, q16, nullptr, nullptr, nullptr);
        gemm(h16, EMB, WkTf + (size_t)l * EMB * EMB, EMB, bk + l * HDIM, HDIM,
             NS, EMB, EMB, 0, k16, nullptr, nullptr, nullptr);
        gemm(h16, EMB, WvTf + (size_t)l * EMB * EMB, EMB, bv + l * HDIM, HDIM,
             NS, EMB, EMB, 0, v16, nullptr, nullptr, nullptr);
        k_vtranspose<<<dim3((unsigned)cdivl(vt_total, 256)), 256, 0, stream>>>(v16, vt16);
        // attention (flash): 8*8*64 q-tiles / 8 waves per block
        k_flash_attn<<<dim3(512), 256, 0, stream>>>(q16, k16, vt16, at16);
        // Wo + residual -> f32
        gemm(at16, EMB, WoT + (size_t)l * EMB * EMB, EMB, bo + l * EMB, EMB,
             NS, EMB, EMB, 3, nullptr, xres, nullptr, h16);
        k_layernorm<<<dim3(NS / 8), 256, 0, stream>>>(xres, g1 + l * EMB, be1 + l * EMB, hx16, NS);
        // FFN
        gemm(hx16, EMB, Wf1T + (size_t)l * EMB * FFD, EMB, bf1 + l * FFD, FFD,
             NS, EMB, FFD, 1, ff16, nullptr, nullptr, nullptr);
        gemm(ff16, FFD, Wf2T + (size_t)l * FFD * EMB, FFD, bf2 + l * EMB, EMB,
             NS, FFD, EMB, 3, nullptr, xres, nullptr, hx16);
        k_layernorm<<<dim3(NS / 8), 256, 0, stream>>>(xres, g2 + l * EMB, be2 + l * EMB, h16, NS);
    }

    // final pointwise conv -> out (N, O, S), transposed store
    gemm(h16, EMB, WfinT, EMB, bfin, OUTC, NS, EMB, OUTC, 4, nullptr, out, nullptr, nullptr);
}
